// EfficientNetSwinUNet_88072599372235
// MI455X (gfx1250) — matrix operand order repaired and checked by reading, not compiled, not run on hardware
//
#include <hip/hip_runtime.h>
#include <cstdint>
#include <cstddef>

// ---------------------------------------------------------------------------
// Swin double-block on gfx1250: bf16 WMMA (v_wmma_f32_16x16x32_bf16) GEMMs,
// wave32 softmax reductions, LDS-staged attention.
// GEMM: 32x64 wave tiles, ping-pong (2x unrolled) register-double-buffered
// K loop (no swap moves), 8 waves/block share one B strip for WGP$/L2 reuse.
// ---------------------------------------------------------------------------

typedef __attribute__((ext_vector_type(16))) __bf16 v16bf;
typedef __attribute__((ext_vector_type(8)))  float  v8f;

#define C_DIM   384
#define C3      1152
#define CH      1536
#define NTOKW   49      // tokens per window
#define NP      64      // padded tokens per window
#define NWIN    1024    // 16 batches * 8 * 8 windows
#define MWIN    65536   // NWIN * NP padded rows
#define MTOK    50176   // 16 * 3136 real tokens
#define HEADS_  12
#define HD_     32
#define SCALE_  0.17677669529663689f  // 32^-0.5

__device__ __forceinline__ __bf16 f2bf(float f) {
  unsigned u = __float_as_uint(f);
  unsigned r = u + 0x7fffu + ((u >> 16) & 1u);   // round-to-nearest-even
  unsigned short h = (unsigned short)(r >> 16);
  return __builtin_bit_cast(__bf16, h);
}

__device__ __forceinline__ int regionOf(int r) { return r < 49 ? 0 : (r < 53 ? 1 : 2); }

// A-matrix 16x32 bf16 tile loader (ISA 7.12.2 16-bit A layout):
// lane lo holds row M=lo; VGPR v pair K = (v&3)*2 + (v>>2)*16 + hi*8.
__device__ __forceinline__ v16bf load_a_tile(const __bf16* __restrict__ base, int rstride) {
  int lane = threadIdx.x & 31;
  int lo = lane & 15, hi = lane >> 4;
  const __bf16* r = base + (size_t)lo * rstride + hi * 8;
  v16bf o;
#pragma unroll
  for (int v = 0; v < 8; ++v) {
    int k = ((v & 3) << 1) + ((v >> 2) << 4);
    o[2 * v]     = r[k];
    o[2 * v + 1] = r[k + 1];
  }
  return o;
}

// B-matrix 32x16 bf16 tile loader: lane lo holds column N=lo;
// VGPR v pair K = (v&3)*2 + (v>>2)*16 + hi*8. Source is N-major (row n contiguous in K).
__device__ __forceinline__ v16bf load_b_tile(const __bf16* __restrict__ base, int rstride) {
  int lane = threadIdx.x & 31;
  int lo = lane & 15, hi = lane >> 4;
  const __bf16* r = base + (size_t)lo * rstride + hi * 8;
  v16bf o;
#pragma unroll
  for (int v = 0; v < 8; ++v) {
    int k = ((v & 3) << 1) + ((v >> 2) << 4);
    o[2 * v]     = r[k];
    o[2 * v + 1] = r[k + 1];
  }
  return o;
}

__device__ __forceinline__ v8f wmma_bf16(v16bf a, v16bf b, v8f c) {
  return __builtin_amdgcn_wmma_f32_16x16x32_bf16(false, a, false, b, (short)0, c, false, false);
}

// --------------------------- weight transpose+convert ----------------------
// src f32 [K,N] row-major  ->  dst bf16 [N,K] row-major (N-major for B loads)
__global__ void wconv_kernel(const float* __restrict__ src, __bf16* __restrict__ dst,
                             int K, int N) {
  size_t tid = (size_t)blockIdx.x * 256 + threadIdx.x;
  if (tid >= (size_t)K * N) return;
  int n = (int)(tid / K);
  int k = (int)(tid % K);
  dst[tid] = f2bf(src[(size_t)k * N + n]);
}

// --------------------------- LN1 + roll + window partition -----------------
// one wave per padded window row; rows >= 49 are zero-filled.
__global__ __launch_bounds__(256) void ln_win_kernel(
    const float* __restrict__ xin, const float* __restrict__ g,
    const float* __restrict__ be, __bf16* __restrict__ xw, int shift) {
  int lane = threadIdx.x & 31;
  int row  = blockIdx.x * 8 + (threadIdx.x >> 5);
  int win  = row >> 6, n = row & 63;
  __bf16* out = xw + (size_t)row * C_DIM;
  if (n >= NTOKW) {
#pragma unroll
    for (int t = 0; t < 12; ++t) out[t * 32 + lane] = f2bf(0.f);
    return;
  }
  int b  = win >> 6;
  int wy = (win >> 3) & 7;
  int wx = win & 7;
  int i = n / 7, j = n % 7;
  int rr = (wy * 7 + i + shift) % 56;
  int cc = (wx * 7 + j + shift) % 56;
  const float* xr = xin + ((size_t)b * 3136 + rr * 56 + cc) * C_DIM;
  float v[12];
  float s = 0.f, ss = 0.f;
#pragma unroll
  for (int t = 0; t < 12; ++t) {
    int c = t * 32 + lane;
    v[t] = xr[c];
    s += v[t];
    ss += v[t] * v[t];
  }
#pragma unroll
  for (int d = 1; d < 32; d <<= 1) {
    s  += __shfl_xor(s,  d, 32);
    ss += __shfl_xor(ss, d, 32);
  }
  float mean = s * (1.f / C_DIM);
  float var  = ss * (1.f / C_DIM) - mean * mean;
  float rs   = rsqrtf(var + 1e-5f);
#pragma unroll
  for (int t = 0; t < 12; ++t) {
    int c = t * 32 + lane;
    out[c] = f2bf((v[t] - mean) * rs * g[c] + be[c]);
  }
}

// --------------------------- LN2 (plain token rows) ------------------------
__global__ __launch_bounds__(256) void ln2_kernel(
    const float* __restrict__ xin, const float* __restrict__ g,
    const float* __restrict__ be, __bf16* __restrict__ out) {
  int lane = threadIdx.x & 31;
  int row  = blockIdx.x * 8 + (threadIdx.x >> 5);
  const float* xr = xin + (size_t)row * C_DIM;
  __bf16* o = out + (size_t)row * C_DIM;
  float v[12];
  float s = 0.f, ss = 0.f;
#pragma unroll
  for (int t = 0; t < 12; ++t) {
    int c = t * 32 + lane;
    v[t] = xr[c];
    s += v[t];
    ss += v[t] * v[t];
  }
#pragma unroll
  for (int d = 1; d < 32; d <<= 1) {
    s  += __shfl_xor(s,  d, 32);
    ss += __shfl_xor(ss, d, 32);
  }
  float mean = s * (1.f / C_DIM);
  float var  = ss * (1.f / C_DIM) - mean * mean;
  float rs   = rsqrtf(var + 1e-5f);
#pragma unroll
  for (int t = 0; t < 12; ++t) {
    int c = t * 32 + lane;
    o[c] = f2bf((v[t] - mean) * rs * g[c] + be[c]);
  }
}

// --------------------------- generic WMMA GEMM -----------------------------
// C[M,Nt] = A_bf16[M,K] @ WT_bf16[Nt,K]^T (+bias, epilogue by mode).
// 8 waves/block; every wave computes a 32x64 strip; all waves of a block use
// the SAME nb (B strip) -> B tiles hit WGP$/L2 8x. K loop unrolled by 2 with
// ping-pong register buffers (K/32 is even for all call sites), so loads for
// one phase overlap WMMAs of the other with no swap moves.
// mode 0: bf16 out = acc+bias              (qkv)
// mode 1: bf16 out = gelu(acc+bias)        (fc1)
// mode 2: f32 out, window-reverse scatter + residual (proj)
// mode 3: f32 out = resid + acc + bias     (fc2)
__global__ __launch_bounds__(256) void gemm_wmma_kernel(
    const __bf16* __restrict__ A, const __bf16* __restrict__ WT,
    const float* __restrict__ bias, void* __restrict__ outp,
    const float* __restrict__ resid, int M, int K, int Nt, int n64,
    int mode, int shift) {
  int w    = threadIdx.x >> 5;
  int nb   = blockIdx.x % n64;
  int mt   = (blockIdx.x / n64) * 8 + w;   // 32-row M tile index
  int lane = threadIdx.x & 31, lo = lane & 15, hi = lane >> 4;

  const __bf16* Ab = A  + (size_t)mt * 32 * K;
  const __bf16* Wb = WT + (size_t)nb * 64 * K;

  v8f acc[8];
#pragma unroll
  for (int t = 0; t < 8; ++t)
#pragma unroll
    for (int r = 0; r < 8; ++r) acc[t][r] = 0.f;

  v16bf ca[2], cb[4], na[2], nbv[4];
  ca[0] = load_a_tile(Ab, K);
  ca[1] = load_a_tile(Ab + (size_t)16 * K, K);
#pragma unroll
  for (int j = 0; j < 4; ++j) cb[j] = load_b_tile(Wb + (size_t)(j * 16) * K, K);

  int nk = K >> 5;                          // even at every call site
  for (int it = 0; it < nk; it += 2) {
    // preload odd phase (it+1 < nk always, nk even)
    int k1 = (it + 1) << 5;
    na[0] = load_a_tile(Ab + k1, K);
    na[1] = load_a_tile(Ab + (size_t)16 * K + k1, K);
#pragma unroll
    for (int j = 0; j < 4; ++j) nbv[j] = load_b_tile(Wb + (size_t)(j * 16) * K + k1, K);

    // even-phase WMMAs
#pragma unroll
    for (int ti = 0; ti < 2; ++ti)
#pragma unroll
      for (int j = 0; j < 4; ++j)
        acc[ti * 4 + j] = wmma_bf16(ca[ti], cb[j], acc[ti * 4 + j]);

    // preload next even phase (uniform branch; EXEC untouched)
    if (it + 2 < nk) {
      int k2 = (it + 2) << 5;
      ca[0] = load_a_tile(Ab + k2, K);
      ca[1] = load_a_tile(Ab + (size_t)16 * K + k2, K);
#pragma unroll
      for (int j = 0; j < 4; ++j) cb[j] = load_b_tile(Wb + (size_t)(j * 16) * K + k2, K);
    }

    // odd-phase WMMAs
#pragma unroll
    for (int ti = 0; ti < 2; ++ti)
#pragma unroll
      for (int j = 0; j < 4; ++j)
        acc[ti * 4 + j] = wmma_bf16(na[ti], nbv[j], acc[ti * 4 + j]);
  }

#pragma unroll
  for (int ti = 0; ti < 2; ++ti) {
#pragma unroll
    for (int j = 0; j < 4; ++j) {
      int n = nb * 64 + j * 16 + lo;
      float bi = bias[n];
#pragma unroll
      for (int r = 0; r < 8; ++r) {
        int m = mt * 32 + ti * 16 + hi * 8 + r;
        float val = acc[ti * 4 + j][r] + bi;
        if (mode == 0) {
          ((__bf16*)outp)[(size_t)m * Nt + n] = f2bf(val);
        } else if (mode == 1) {
          float gl = 0.5f * val * (1.f + erff(val * 0.70710678118654752f));
          ((__bf16*)outp)[(size_t)m * Nt + n] = f2bf(gl);
        } else if (mode == 2) {
          int niw = m & 63;
          if (niw < NTOKW) {
            int win = m >> 6;
            int b  = win >> 6;
            int wy = (win >> 3) & 7;
            int wx = win & 7;
            int i = niw / 7, jj = niw % 7;
            int rr = (wy * 7 + i + shift) % 56;
            int cc = (wx * 7 + jj + shift) % 56;
            size_t t = ((size_t)b * 3136 + rr * 56 + cc) * C_DIM + n;
            ((float*)outp)[t] = resid[t] + val;
          }
        } else {  // mode 3
          size_t t = (size_t)m * C_DIM + n;
          ((float*)outp)[t] = resid[t] + val;
        }
      }
    }
  }
}

// --------------------------- attention per (window, head) ------------------
// 4 waves / 128 threads. S = Q K^T (1 K-step of 32), softmax (shfl_xor row
// reductions over 16-lane halves), P staged in LDS -> A layout, V^T staged in
// LDS -> B layout, O = P V (2 K-steps).
__global__ __launch_bounds__(128) void attn_kernel(
    const __bf16* __restrict__ qkv, const float* __restrict__ rpb,
    __bf16* __restrict__ o, int shift) {
  __shared__ __bf16 vT[32 * 64];   // dim-major V: vT[d*64 + tok]
  __shared__ __bf16 pS[64 * 64];   // softmax probs, row-major

  int win  = blockIdx.x / HEADS_;
  int head = blockIdx.x % HEADS_;
  int wid  = threadIdx.x >> 5;
  int lane = threadIdx.x & 31, lo = lane & 15, hi = lane >> 4;

  // stage V^T
  for (int idx = threadIdx.x; idx < 2048; idx += 128) {
    int tok = idx >> 5, d = idx & 31;
    vT[d * 64 + tok] = qkv[(size_t)(win * 64 + tok) * C3 + 2 * C_DIM + head * HD_ + d];
  }
  __syncthreads();

  // scores for this wave's 16 rows
  const __bf16* qb = qkv + (size_t)(win * 64 + wid * 16) * C3 + head * HD_;
  v16bf aq = load_a_tile(qb, C3);
  const __bf16* kb = qkv + (size_t)(win * 64) * C3 + C_DIM + head * HD_;
  v8f s[4];
#pragma unroll
  for (int j = 0; j < 4; ++j) {
    v8f z;
#pragma unroll
    for (int r = 0; r < 8; ++r) z[r] = 0.f;
    v16bf bk = load_b_tile(kb + (size_t)(j * 16) * C3, C3);
    s[j] = wmma_bf16(aq, bk, z);
  }

  int wy = (win >> 3) & 7, wx = win & 7;
#pragma unroll
  for (int r = 0; r < 8; ++r) {
    int m = wid * 16 + hi * 8 + r;
    bool mv = m < NTOKW;
    int i1 = m / 7, j1 = m % 7;
    int gn = 0;
    if (shift && mv) gn = regionOf(wy * 7 + i1) * 3 + regionOf(wx * 7 + j1);
    float lv[4];
    float mx = -1e30f;
#pragma unroll
    for (int j = 0; j < 4; ++j) {
      int nc = j * 16 + lo;
      float x;
      if (!mv) {
        x = 0.f;                         // padded row: uniform, discarded later
      } else if (nc >= NTOKW) {
        x = -1e30f;                      // padded key column
      } else {
        int i2 = nc / 7, j2 = nc % 7;
        int ridx = (i1 - i2 + 6) * 13 + (j1 - j2 + 6);
        x = s[j][r] * SCALE_ + rpb[ridx * HEADS_ + head];
        if (shift) {
          int gm = regionOf(wy * 7 + i2) * 3 + regionOf(wx * 7 + j2);
          if (gm != gn) x -= 100.f;
        }
      }
      lv[j] = x;
      mx = fmaxf(mx, x);
    }
#pragma unroll
    for (int d = 1; d < 16; d <<= 1) mx = fmaxf(mx, __shfl_xor(mx, d, 32));
    float sum = 0.f;
#pragma unroll
    for (int j = 0; j < 4; ++j) {
      lv[j] = __expf(lv[j] - mx);
      sum += lv[j];
    }
#pragma unroll
    for (int d = 1; d < 16; d <<= 1) sum += __shfl_xor(sum, d, 32);
    float inv = 1.f / sum;
#pragma unroll
    for (int j = 0; j < 4; ++j) pS[m * 64 + j * 16 + lo] = f2bf(lv[j] * inv);
  }
  __syncthreads();

  // O = P @ V
  v8f ov[2];
#pragma unroll
  for (int j2 = 0; j2 < 2; ++j2)
#pragma unroll
    for (int r = 0; r < 8; ++r) ov[j2][r] = 0.f;
  for (int kk = 0; kk < 64; kk += 32) {
    v16bf pa = load_a_tile(&pS[(wid * 16) * 64 + kk], 64);
#pragma unroll
    for (int j2 = 0; j2 < 2; ++j2) {
      v16bf vb = load_b_tile(&vT[(j2 * 16) * 64 + kk], 64);
      ov[j2] = wmma_bf16(pa, vb, ov[j2]);
    }
  }
#pragma unroll
  for (int j2 = 0; j2 < 2; ++j2)
#pragma unroll
    for (int r = 0; r < 8; ++r) {
      size_t row = (size_t)(win * 64 + wid * 16 + hi * 8 + r);
      o[row * C_DIM + head * HD_ + j2 * 16 + lo] = f2bf(ov[j2][r]);
    }
}

// --------------------------- orchestration ---------------------------------
extern "C" void kernel_launch(void* const* d_in, const int* in_sizes, int n_in,
                              void* d_out, int out_size, void* d_ws, size_t ws_size,
                              hipStream_t stream) {
  (void)in_sizes; (void)n_in; (void)out_size; (void)ws_size;

  const float* x_in  = (const float*)d_in[0];
  const float* n1g   = (const float*)d_in[1];
  const float* n1b   = (const float*)d_in[2];
  const float* qkvw  = (const float*)d_in[3];
  const float* qkvb  = (const float*)d_in[4];
  const float* rpb   = (const float*)d_in[5];
  const float* projw = (const float*)d_in[6];
  const float* projb = (const float*)d_in[7];
  const float* n2g   = (const float*)d_in[8];
  const float* n2b   = (const float*)d_in[9];
  const float* fc1w  = (const float*)d_in[10];
  const float* fc1b  = (const float*)d_in[11];
  const float* fc2w  = (const float*)d_in[12];
  const float* fc2b  = (const float*)d_in[13];

  // workspace carve-up (256B aligned)
  uintptr_t p = (uintptr_t)d_ws;
  auto carve = [&](size_t bytes) -> void* {
    void* r = (void*)p;
    p += (bytes + 255) & ~(size_t)255;
    return r;
  };
  __bf16* wT   = (__bf16*)carve((size_t)3538944 * 2);          // transposed weights
  __bf16* bufA = (__bf16*)carve((size_t)MWIN * C_DIM * 2);     // xw / o / ln2
  __bf16* bufB = (__bf16*)carve((size_t)MTOK * CH * 2);        // qkv / fc1-hidden
  float*  xbuf = (float*) carve((size_t)MTOK * C_DIM * 4);     // running x

  __bf16* wT_qkv  = wT;
  __bf16* wT_proj = wT + 884736;
  __bf16* wT_fc1  = wT + 1179648;
  __bf16* wT_fc2  = wT + 2359296;

  auto wconv = [&](const float* s, __bf16* d, int K, int N) {
    size_t tot = (size_t)K * N;
    wconv_kernel<<<(unsigned)((tot + 255) / 256), 256, 0, stream>>>(s, d, K, N);
  };
  wconv(qkvw,            wT_qkv,            C_DIM, C3);
  wconv(qkvw + 442368,   wT_qkv + 442368,   C_DIM, C3);
  wconv(projw,           wT_proj,           C_DIM, C_DIM);
  wconv(projw + 147456,  wT_proj + 147456,  C_DIM, C_DIM);
  wconv(fc1w,            wT_fc1,            C_DIM, CH);
  wconv(fc1w + 589824,   wT_fc1 + 589824,   C_DIM, CH);
  wconv(fc2w,            wT_fc2,            CH, C_DIM);
  wconv(fc2w + 589824,   wT_fc2 + 589824,   CH, C_DIM);

  for (int blk = 0; blk < 2; ++blk) {
    int shift = blk ? 3 : 0;
    const float* xin  = blk ? xbuf : x_in;
    float*       xout = blk ? (float*)d_out : xbuf;

    // LN1 + roll + window partition -> bufA (bf16, padded windows)
    ln_win_kernel<<<MWIN / 8, 256, 0, stream>>>(
        xin, n1g + blk * C_DIM, n1b + blk * C_DIM, bufA, shift);

    // QKV: [65536,384] x [384,1152]; 2048 M-tiles(32) / 8 waves * 18 N-strips
    gemm_wmma_kernel<<<(2048 / 8) * 18, 256, 0, stream>>>(
        bufA, wT_qkv + blk * 442368, qkvb + blk * C3, (void*)bufB, nullptr,
        MWIN, C_DIM, C3, 18, 0, 0);

    // attention -> bufA (reused as O)
    attn_kernel<<<NWIN * HEADS_, 128, 0, stream>>>(
        bufB, rpb + blk * 169 * HEADS_, bufA, shift);

    // proj + window reverse + roll back + residual -> xbuf (f32)
    gemm_wmma_kernel<<<(2048 / 8) * 6, 256, 0, stream>>>(
        bufA, wT_proj + blk * 147456, projb + blk * C_DIM, (void*)xbuf, xin,
        MWIN, C_DIM, C_DIM, 6, 2, shift);

    // LN2 -> bufA (bf16, first MTOK rows)
    ln2_kernel<<<MTOK / 8, 256, 0, stream>>>(
        xbuf, n2g + blk * C_DIM, n2b + blk * C_DIM, bufA);

    // fc1 + GELU: [50176,384] x [384,1536]; 1568 M-tiles / 8 * 24 N-strips
    gemm_wmma_kernel<<<(1568 / 8) * 24, 256, 0, stream>>>(
        bufA, wT_fc1 + blk * 589824, fc1b + blk * CH, (void*)bufB, nullptr,
        MTOK, C_DIM, CH, 24, 1, 0);

    // fc2 + residual: [50176,1536] x [1536,384]
    gemm_wmma_kernel<<<(1568 / 8) * 6, 256, 0, stream>>>(
        bufB, wT_fc2 + blk * 589824, fc2b + blk * C_DIM, (void*)xout, xbuf,
        MTOK, CH, C_DIM, 6, 3, 0);
  }
}
